// SaliencyMB_SS2D_9483287790158
// MI455X (gfx1250) — compile-verified, matrix operand order checked by reading, not run on hardware
//
#include <hip/hip_runtime.h>
#include <hip/hip_bf16.h>
#include <math.h>

typedef __bf16 bf16_t;
typedef __attribute__((ext_vector_type(16))) __bf16 v16bf;
typedef __attribute__((ext_vector_type(8)))  __bf16 v8bf;
typedef __attribute__((ext_vector_type(8)))  float  v8f;

// Fixed problem dims from setup_inputs()
#define BB 8
#define HH 64
#define WW 64
#define CC 96
#define DD 192
#define NN 4
#define RR 6
#define KK 4
#define LL 4096          // H*W
#define MROWS 32768      // B*L

// ---------------- f32 -> bf16 conversion ----------------
__global__ void k_cvt_bf16(const float* __restrict__ in, bf16_t* __restrict__ out, int n) {
  int i = blockIdx.x * blockDim.x + threadIdx.x;
  if (i < n) out[i] = (bf16_t)in[i];
}

// ---------------- WMMA bf16 GEMM: C[M,N] = A[M,K] * W[N,K]^T ----------------
// One wave per 16x16 output tile. A,W row-major bf16; C row-major f32.
// Per ISA 16-bit A layout (wave32): lane = m + 16*g; elems 0..7 -> K=g*8..g*8+7,
// elems 8..15 -> K=16+g*8..16+g*8+7. B symmetric with N per lane. C/D: lane n+16g,
// vgpr r -> D[r+8g][n].
__global__ void k_wmma_gemm(const bf16_t* __restrict__ A, const bf16_t* __restrict__ W,
                            float* __restrict__ Cout, int M, int Nn, int Kk) {
  const int wid  = (blockIdx.x * blockDim.x + threadIdx.x) >> 5;
  const int lane = threadIdx.x & 31;
  const int tilesN = Nn >> 4;
  const int tiles  = (M >> 4) * tilesN;
  if (wid >= tiles) return;                 // wave-uniform: EXEC stays all-ones
  const int tm = wid / tilesN, tn = wid % tilesN;
  const int g   = lane >> 4;
  const int r16 = lane & 15;
  const bf16_t* arow = A + (size_t)(tm * 16 + r16) * Kk;
  const bf16_t* wrow = W + (size_t)(tn * 16 + r16) * Kk;
  v8f acc = {};
  for (int k0 = 0; k0 < Kk; k0 += 32) {
    v8bf alo = *(const v8bf*)(arow + k0 + 8 * g);
    v8bf ahi = *(const v8bf*)(arow + k0 + 16 + 8 * g);
    v8bf blo = *(const v8bf*)(wrow + k0 + 8 * g);
    v8bf bhi = *(const v8bf*)(wrow + k0 + 16 + 8 * g);
    v16bf a, b;
#pragma unroll
    for (int e = 0; e < 8; ++e) {
      a[e] = alo[e]; a[e + 8] = ahi[e];
      b[e] = blo[e]; b[e + 8] = bhi[e];
    }
    acc = __builtin_amdgcn_wmma_f32_16x16x32_bf16(false, a, false, b,
                                                  (short)0, acc, false, false);
  }
#pragma unroll
  for (int r = 0; r < 8; ++r)
    Cout[(size_t)(tm * 16 + r + 8 * g) * Nn + tn * 16 + r16] = acc[r];
}

// ---------------- depthwise 3x3 conv + SiLU:  xh(B,H,W,D) -> xc(B,L,D) ----------------
__global__ void k_conv_silu(const float* __restrict__ xh, const float* __restrict__ cw,
                            const float* __restrict__ cb, float* __restrict__ xc) {
  const int b = blockIdx.x / HH, h = blockIdx.x % HH, d = threadIdx.x;
  float w9[9];
#pragma unroll
  for (int t = 0; t < 9; ++t) w9[t] = cw[d * 9 + t];
  const float bias = cb[d];
  for (int wc = 0; wc < WW; ++wc) {
    float s = bias;
#pragma unroll
    for (int dh = -1; dh <= 1; ++dh) {
      const int hh = h + dh;
      if ((unsigned)hh >= (unsigned)HH) continue;
#pragma unroll
      for (int dw = -1; dw <= 1; ++dw) {
        const int wp = wc + dw;
        if ((unsigned)wp >= (unsigned)WW) continue;
        s += xh[(((size_t)b * HH + hh) * WW + wp) * DD + d] * w9[(dh + 1) * 3 + (dw + 1)];
      }
    }
    s = s / (1.f + __expf(-s));   // SiLU
    xc[(((size_t)b * HH + h) * WW + wc) * DD + d] = s;
  }
}

// ---------------- per-(batch,layout) bitonic argsort of saliency; emit idxmap ----------------
// layout 0: HW-major (raster); layout 1: WH-major. idxmap[b][k][t] = raster index that
// scan direction k touches at scan step t (same index for gather and scatter).
__global__ void k_sort_idx(const float* __restrict__ gt, int* __restrict__ idxmap) {
  __shared__ float key[LL];
  __shared__ unsigned short ridx[LL];
  const int b = blockIdx.x >> 1, layout = blockIdx.x & 1;
  const float* g = gt + (size_t)b * LL;
  for (int i = threadIdx.x; i < LL; i += blockDim.x) {
    const int raster = layout ? ((i & 63) * 64 + (i >> 6)) : i;
    key[i] = -g[raster];                 // ascending sort of -gt == descending gt
    ridx[i] = (unsigned short)i;
  }
  __syncthreads();
  for (int ks = 2; ks <= LL; ks <<= 1) {
    for (int j = ks >> 1; j > 0; j >>= 1) {
      for (int i = threadIdx.x; i < LL; i += blockDim.x) {
        const int p = i ^ j;
        if (p > i) {
          const bool up = ((i & ks) == 0);
          const float ki = key[i], kp = key[p];
          if ((ki > kp) == up) {
            key[i] = kp; key[p] = ki;
            unsigned short t = ridx[i]; ridx[i] = ridx[p]; ridx[p] = t;
          }
        }
      }
      __syncthreads();
    }
  }
  int* m0 = idxmap + ((size_t)b * KK + layout) * LL;       // forward direction
  int* m2 = idxmap + ((size_t)b * KK + layout + 2) * LL;   // flipped direction
  for (int t = threadIdx.x; t < LL; t += blockDim.x) {
    const int o = ridx[t];
    const int raster = layout ? ((o & 63) * 64 + (o >> 6)) : o;
    m0[t] = raster;
    m2[LL - 1 - t] = raster;
  }
}

// ---------------- x_dbl: dt(R)->softplus(dt_proj) and B,C per (b,k,l) ----------------
#define TLX 16
__global__ void k_xdbl(const float* __restrict__ xc, const int* __restrict__ idxmap,
                       const float* __restrict__ xpw, const float* __restrict__ dtw,
                       const float* __restrict__ dtb,
                       float* __restrict__ dts, float* __restrict__ bscs) {
  __shared__ float xt[TLX][DD];
  __shared__ float dtr[TLX][RR];
  __shared__ int   sidx[TLX];
  const int tile = blockIdx.x % (LL / TLX);
  const int bk   = blockIdx.x / (LL / TLX);
  const int b = bk / KK, k = bk % KK;
  const int l0 = tile * TLX;
  const int tid = threadIdx.x;
  if (tid < TLX) sidx[tid] = idxmap[(size_t)bk * LL + l0 + tid];
  __syncthreads();
  for (int i = tid; i < TLX * DD; i += blockDim.x) {
    const int lt = i / DD, d = i % DD;
    xt[lt][d] = xc[((size_t)b * LL + sidx[lt]) * DD + d];   // gather via L2-resident xc
  }
  __syncthreads();
  if (tid < TLX * (RR + 2 * NN)) {
    const int lt = tid / (RR + 2 * NN), c = tid % (RR + 2 * NN);
    const float* wrow = xpw + ((size_t)k * (RR + 2 * NN) + c) * DD;
    float s = 0.f;
    for (int d = 0; d < DD; ++d) s += xt[lt][d] * wrow[d];
    if (c < RR) dtr[lt][c] = s;
    else        bscs[((size_t)bk * LL + l0 + lt) * (2 * NN) + (c - RR)] = s;
  }
  __syncthreads();
  for (int i = tid; i < TLX * DD; i += blockDim.x) {
    const int lt = i / DD, d = i % DD;
    float s = dtb[k * DD + d];
    const float* wr = dtw + ((size_t)k * DD + d) * RR;
#pragma unroll
    for (int r = 0; r < RR; ++r) s += wr[r] * dtr[lt][r];
    s = (s > 20.f) ? s : log1pf(__expf(s));                 // softplus
    dts[((size_t)bk * LL + l0 + lt) * DD + d] = s;
  }
}

// ---------------- selective scan, one workgroup per (b,k), deterministic scatter ----------------
#define SCHUNK 64
__global__ void k_scan(const float* __restrict__ xc, const float* __restrict__ dts,
                       const float* __restrict__ bscs, const int* __restrict__ idxmap,
                       const float* __restrict__ A_logs, const float* __restrict__ Dsv,
                       float* __restrict__ ysc) {
  const int bk = blockIdx.x, b = bk / KK, k = bk % KK;
  const int d = threadIdx.x;     // DD = 192 threads = 6 wave32s
  float Av[NN], h[NN];
#pragma unroll
  for (int n = 0; n < NN; ++n) {
    Av[n] = -__expf(A_logs[((size_t)k * DD + d) * NN + n]);
    h[n] = 0.f;
  }
  const float Dd = Dsv[k * DD + d];
  __shared__ float bc[SCHUNK][2 * NN];
  __shared__ int   sidx[SCHUNK];
  const int*   im    = idxmap + (size_t)bk * LL;
  const float* dtrow = dts  + (size_t)bk * LL * DD;
  const float* bcrow = bscs + (size_t)bk * LL * (2 * NN);
  const float* xcb   = xc   + (size_t)b * LL * DD;
  float*       yrow  = ysc  + ((size_t)k * BB + b) * LL * DD;
  for (int l0 = 0; l0 < LL; l0 += SCHUNK) {
    __syncthreads();
    for (int i = d; i < SCHUNK * 2 * NN; i += DD)
      ((float*)bc)[i] = bcrow[(size_t)l0 * 2 * NN + i];
    for (int i = d; i < SCHUNK; i += DD) sidx[i] = im[l0 + i];
    __syncthreads();
    if (l0 + SCHUNK < LL)  // hint next dts chunk toward the caches (global_prefetch_b8)
      __builtin_prefetch(&dtrow[(size_t)(l0 + SCHUNK) * DD + d], 0, 1);
    for (int t = 0; t < SCHUNK; ++t) {
      const int   raster = sidx[t];
      const float u   = xcb[(size_t)raster * DD + d];
      const float dtv = dtrow[(size_t)(l0 + t) * DD + d];
      const float du  = dtv * u;
      float y = u * Dd;
#pragma unroll
      for (int n = 0; n < NN; ++n) {
        const float dA = __expf(dtv * Av[n]);
        h[n] = fmaf(dA, h[n], du * bc[t][n]);
        y    = fmaf(h[n], bc[t][NN + n], y);
      }
      yrow[(size_t)raster * DD + d] = y;   // scatter to raster order (permutation: no races)
    }
  }
}

// ---------------- squeeze (mean over L) ----------------
__global__ void k_sq(const float* __restrict__ xc, float* __restrict__ sq) {
  const int b = blockIdx.x, d = threadIdx.x;
  float s = 0.f;
  for (int l = 0; l < LL; ++l) s += xc[((size_t)b * LL + l) * DD + d];
  sq[b * DD + d] = s * (1.f / LL);
}

// ---------------- excitation gate ----------------
__global__ void k_se(const float* __restrict__ sq, const float* __restrict__ fc1,
                     const float* __restrict__ fc2, float* __restrict__ ex) {
  __shared__ float sv[DD];
  __shared__ float h1[DD / 16];
  const int b = blockIdx.x, tid = threadIdx.x;
  sv[tid] = sq[b * DD + tid];
  __syncthreads();
  if (tid < DD / 16) {
    float s = 0.f;
    for (int d = 0; d < DD; ++d) s += fc1[tid * DD + d] * sv[d];
    h1[tid] = s / (1.f + __expf(-s));          // SiLU
  }
  __syncthreads();
  float s = 0.f;
#pragma unroll
  for (int r = 0; r < DD / 16; ++r) s += fc2[tid * (DD / 16) + r] * h1[r];
  ex[b * DD + tid] = 1.f / (1.f + __expf(-s)); // sigmoid
}

// ---------------- merge 4 directions + LayerNorm + gate -> bf16 for out_proj ----------------
__global__ void k_ln_gate(const float* __restrict__ ysc, const float* __restrict__ gamma,
                          const float* __restrict__ beta, const float* __restrict__ ex,
                          bf16_t* __restrict__ out) {
  const int row  = blockIdx.x * 8 + (threadIdx.x >> 5);   // b*LL + l, one wave per row
  const int lane = threadIdx.x & 31;
  const int b = row >> 12;
  float v[6];
  float s = 0.f;
#pragma unroll
  for (int i = 0; i < 6; ++i) {
    const int d = lane + 32 * i;
    float acc = 0.f;
#pragma unroll
    for (int k = 0; k < KK; ++k)
      acc += ysc[((size_t)k * BB * LL + row) * DD + d];
    v[i] = acc; s += acc;
  }
#pragma unroll
  for (int o = 16; o > 0; o >>= 1) s += __shfl_xor(s, o, 32);
  const float m = s * (1.f / DD);
  float vs = 0.f;
#pragma unroll
  for (int i = 0; i < 6; ++i) { const float dv = v[i] - m; vs += dv * dv; }
#pragma unroll
  for (int o = 16; o > 0; o >>= 1) vs += __shfl_xor(vs, o, 32);
  const float rstd = rsqrtf(vs * (1.f / DD) + 1e-5f);
  const float* exb = ex + b * DD;
#pragma unroll
  for (int i = 0; i < 6; ++i) {
    const int d = lane + 32 * i;
    out[(size_t)row * DD + d] =
        (bf16_t)(((v[i] - m) * rstd * gamma[d] + beta[d]) * exb[d]);
  }
}

// ---------------- host launcher ----------------
extern "C" void kernel_launch(void* const* d_in, const int* in_sizes, int n_in,
                              void* d_out, int out_size, void* d_ws, size_t ws_size,
                              hipStream_t stream) {
  (void)in_sizes; (void)n_in; (void)out_size; (void)ws_size;
  const float* x    = (const float*)d_in[0];
  const float* gt   = (const float*)d_in[1];
  const float* win  = (const float*)d_in[2];
  const float* cw   = (const float*)d_in[3];
  const float* cb   = (const float*)d_in[4];
  const float* xpw  = (const float*)d_in[5];
  const float* dtw  = (const float*)d_in[6];
  const float* dtb  = (const float*)d_in[7];
  const float* alog = (const float*)d_in[8];
  const float* dsv  = (const float*)d_in[9];
  const float* gam  = (const float*)d_in[10];
  const float* bet  = (const float*)d_in[11];
  const float* fc1  = (const float*)d_in[12];
  const float* fc2  = (const float*)d_in[13];
  const float* wout = (const float*)d_in[14];
  float* out = (float*)d_out;
  char*  ws  = (char*)d_ws;

  size_t off = 0;
  auto alloc = [&](size_t bytes) {
    size_t o = off; off = (off + bytes + 255) & ~(size_t)255; return o;
  };
  const size_t oXH   = alloc((size_t)MROWS * DD * 4);          // in_proj output (B,L,D)
  const size_t oXC   = alloc((size_t)MROWS * DD * 4);          // conv+silu (B,L,D) — 25MB, L2-resident
  const size_t oXBF  = alloc((size_t)MROWS * CC * 2);          // x as bf16
  const size_t oWIN  = alloc((size_t)DD * CC * 2);             // in_proj_w bf16
  const size_t oWOUT = alloc((size_t)CC * DD * 2);             // out_proj_w bf16
  const size_t oIDX  = alloc((size_t)BB * KK * LL * 4);        // scan index maps
  const size_t oDTS  = alloc((size_t)BB * KK * LL * DD * 4);   // softplus(dt)
  const size_t oBC   = alloc((size_t)BB * KK * LL * 2 * NN * 4); // Bs|Cs
  const size_t oYSC  = alloc((size_t)KK * BB * LL * DD * 4);   // per-direction raster scan out
  const size_t oYLN  = alloc((size_t)MROWS * DD * 2);          // LN+gate output bf16
  const size_t oSQ   = alloc((size_t)BB * DD * 4);
  const size_t oEX   = alloc((size_t)BB * DD * 4);

  float*  xh    = (float*)(ws + oXH);
  float*  xcbuf = (float*)(ws + oXC);
  bf16_t* xbf   = (bf16_t*)(ws + oXBF);
  bf16_t* winbf = (bf16_t*)(ws + oWIN);
  bf16_t* woutbf= (bf16_t*)(ws + oWOUT);
  int*    idxm  = (int*)(ws + oIDX);
  float*  dtsb  = (float*)(ws + oDTS);
  float*  bcb   = (float*)(ws + oBC);
  float*  yscb  = (float*)(ws + oYSC);
  bf16_t* ylnbf = (bf16_t*)(ws + oYLN);
  float*  sqb   = (float*)(ws + oSQ);
  float*  exb   = (float*)(ws + oEX);

  // 1) bf16 conversions
  {
    const int n = MROWS * CC;
    k_cvt_bf16<<<(n + 255) / 256, 256, 0, stream>>>(x, xbf, n);
  }
  k_cvt_bf16<<<(DD * CC + 255) / 256, 256, 0, stream>>>(win, winbf, DD * CC);
  k_cvt_bf16<<<(DD * CC + 255) / 256, 256, 0, stream>>>(wout, woutbf, DD * CC);

  // 2) in_proj GEMM (WMMA bf16): (32768 x 96) x (192 x 96)^T -> (32768 x 192)
  k_wmma_gemm<<<(MROWS / 16) * (DD / 16) / 8, 256, 0, stream>>>(xbf, winbf, xh,
                                                                MROWS, DD, CC);
  // 3) depthwise conv + SiLU
  k_conv_silu<<<BB * HH, DD, 0, stream>>>(xh, cw, cb, xcbuf);
  // 4) saliency bitonic sorts -> idx maps (covers all 4 directions)
  k_sort_idx<<<BB * 2, 256, 0, stream>>>(gt, idxm);
  // 5) dt / B / C projections
  k_xdbl<<<BB * KK * (LL / TLX), 256, 0, stream>>>(xcbuf, idxm, xpw, dtw, dtb, dtsb, bcb);
  // 6) selective scan with raster scatter
  k_scan<<<BB * KK, DD, 0, stream>>>(xcbuf, dtsb, bcb, idxm, alog, dsv, yscb);
  // 7) squeeze-excitation gate
  k_sq<<<BB, DD, 0, stream>>>(xcbuf, sqb);
  k_se<<<BB, DD, 0, stream>>>(sqb, fc1, fc2, exb);
  // 8) merge + LayerNorm + gate -> bf16
  k_ln_gate<<<MROWS / 8, 256, 0, stream>>>(yscb, gam, bet, exb, ylnbf);
  // 9) out_proj GEMM (WMMA bf16): (32768 x 192) x (96 x 192)^T -> d_out (32768 x 96)
  k_wmma_gemm<<<(MROWS / 16) * (CC / 16) / 8, 256, 0, stream>>>(ylnbf, woutbf, out,
                                                                MROWS, CC, DD);
}
